// MAHGN_43112881717516
// MI455X (gfx1250) — matrix operand order repaired and checked by reading, not compiled, not run on hardware
//
#include <hip/hip_runtime.h>

typedef __attribute__((ext_vector_type(16))) _Float16 v16h;
typedef __attribute__((ext_vector_type(8)))  float    v8f;

#define NU 200000
#define NA 100000
#define NC 500

__device__ __forceinline__ float lrelu(float x, float s) { return x > 0.f ? x : s * x; }

// float atomic max via signed-max / unsigned-min trick (handles negatives, init = -inf)
__device__ __forceinline__ void atomicMaxF(float* addr, float val) {
  if (val >= 0.f) atomicMax((int*)addr, __float_as_int(val));
  else            atomicMin((unsigned int*)addr, (unsigned int)__float_as_int(val));
}

// ---------------------------------------------------------------------------
// Y[N,64] = X[N,64] @ W[64,64]  (f16 operands, f32 accumulate via WMMA)
// alpha[N,4] = sum_c Y[n, h*16+c] * att[h,c]   (fused from WMMA C fragments)
// block = 128 threads = 4 waves; each wave owns a 16-row strip.
// W is staged in LDS pre-swizzled into the per-lane B-fragment layout so each
// fragment is one contiguous 32B per-lane read (2x ds_load_b128, no packing).
// Out-of-range A rows are clamped (not zeroed): row m of D depends only on
// row m of A, and padded rows are masked at the stores, so no selects needed.
// ---------------------------------------------------------------------------
__global__ __launch_bounds__(128)
void hgat_xform(const float* __restrict__ X, const float* __restrict__ W,
                const float* __restrict__ att, float* __restrict__ Y,
                float* __restrict__ alpha, int N, int storeY) {
  // Wh[kc][tile][lane][j] = W[k][col], k = kc*32 + (lane>=16)*16 + j,
  //                                  col = tile*16 + (lane&15)
  __shared__ __align__(16) _Float16 Wh[2][4][32][16];
  const int tid = threadIdx.x;
  #pragma unroll 8
  for (int i = tid; i < 64 * 64; i += 128) {
    const int k = i >> 6, col = i & 63;
    const int kc = k >> 5, kk = k & 31;
    Wh[kc][col >> 4][(kk >> 4) * 16 + (col & 15)][kk & 15] = (_Float16)W[i];
  }
  __syncthreads();

  const int lane = tid & 31;
  const int wave = tid >> 5;
  const int l15  = lane & 15;
  const int hi   = (lane >> 4) & 1;            // half-wave select
  const int rowBase = blockIdx.x * 64 + wave * 16;

  // --- A fragments (16x32 f16, two K chunks), vectorized loads ------------
  const int rowA  = rowBase + l15;
  const int rowAc = rowA < N ? rowA : (N - 1);  // clamp; padded rows masked at store
  const float* xrow = X + (size_t)rowAc * 64;
  v16h aF[2];
  #pragma unroll
  for (int kc = 0; kc < 2; ++kc) {
    const int kb = kc * 32 + hi * 16;
    const float4 r0 = *(const float4*)(xrow + kb + 0);
    const float4 r1 = *(const float4*)(xrow + kb + 4);
    const float4 r2 = *(const float4*)(xrow + kb + 8);
    const float4 r3 = *(const float4*)(xrow + kb + 12);
    const float fv[16] = { r0.x, r0.y, r0.z, r0.w, r1.x, r1.y, r1.z, r1.w,
                           r2.x, r2.y, r2.z, r2.w, r3.x, r3.y, r3.z, r3.w };
    #pragma unroll
    for (int j = 0; j < 16; ++j)
      aF[kc][j] = (_Float16)fv[j];
  }

  // --- B fragments: contiguous per-lane reads from swizzled LDS -----------
  v16h bF[2][4];
  #pragma unroll
  for (int kc = 0; kc < 2; ++kc)
    #pragma unroll
    for (int t = 0; t < 4; ++t)
      bF[kc][t] = *(const v16h*)&Wh[kc][t][lane][0];

  // --- 8x WMMA: 16x64 strip -----------------------------------------------
  v8f acc[4] = { {}, {}, {}, {} };
  #pragma unroll
  for (int kc = 0; kc < 2; ++kc) {
    #pragma unroll
    for (int t = 0; t < 4; ++t)
      acc[t] = __builtin_amdgcn_wmma_f32_16x16x32_f16(
          false, aF[kc], false, bF[kc][t], (short)0, acc[t], false, false);
  }

  // --- store Y and fused attention logits ---------------------------------
  float attv[4];
  #pragma unroll
  for (int t = 0; t < 4; ++t) attv[t] = att[t * 16 + l15];

  #pragma unroll
  for (int r = 0; r < 8; ++r) {
    const int  rowM = rowBase + hi * 8 + r;
    const bool ok   = rowM < N;
    #pragma unroll
    for (int t = 0; t < 4; ++t) {
      const float v = acc[t][r];
      if (storeY && ok) Y[(size_t)rowM * 64 + t * 16 + l15] = v;
      float p = v * attv[t];                  // head t == column tile t (C == 16)
      p += __shfl_xor(p, 1, 32);
      p += __shfl_xor(p, 2, 32);
      p += __shfl_xor(p, 4, 32);
      p += __shfl_xor(p, 8, 32);
      if (ok && l15 == 0) alpha[(size_t)rowM * 4 + t] = p;
    }
  }
}

// ---------------------------------------------------------------------------
__global__ void hgat_copy_init(const float* __restrict__ src, float* __restrict__ dst,
                               float* __restrict__ sum, int n) {
  int i = blockIdx.x * blockDim.x + threadIdx.x;
  if (i < n) { float v = src[i]; dst[i] = v; if (sum) sum[i] = v; }
}

__global__ void hgat_init_md(float* __restrict__ m, float* __restrict__ den, int n) {
  int i = blockIdx.x * blockDim.x + threadIdx.x;
  if (i < n) { m[i] = -INFINITY; den[i] = 0.f; }
}

__global__ void hgat_out_bias(float* __restrict__ out, const float* __restrict__ biasL,
                              int t0, int t1, int t2, int t3, int cnt, int n) {
  int i = blockIdx.x * blockDim.x + threadIdx.x;
  if (i >= n) return;
  int f = i & 63;
  float s = biasL[t0 * 64 + f];
  if (cnt > 1) s += biasL[t1 * 64 + f];
  if (cnt > 2) s += biasL[t2 * 64 + f];
  if (cnt > 3) s += biasL[t3 * 64 + f];
  out[i] = s;
}

__global__ void hgat_edge_max(const int* __restrict__ ei, int E,
                              const float* __restrict__ aS, const float* __restrict__ aD,
                              float* __restrict__ m) {
  int e = blockIdx.x * blockDim.x + threadIdx.x;
  if (e >= E) return;
  int row = ei[e], col = ei[E + e];
  const float4 s4 = *(const float4*)(aS + (size_t)row * 4);
  const float4 d4 = *(const float4*)(aD + (size_t)col * 4);
  float* mp = m + (size_t)col * 4;
  atomicMaxF(mp + 0, lrelu(s4.x + d4.x, 0.2f));
  atomicMaxF(mp + 1, lrelu(s4.y + d4.y, 0.2f));
  atomicMaxF(mp + 2, lrelu(s4.z + d4.z, 0.2f));
  atomicMaxF(mp + 3, lrelu(s4.w + d4.w, 0.2f));
}

__global__ void hgat_edge_den(const int* __restrict__ ei, int E,
                              const float* __restrict__ aS, const float* __restrict__ aD,
                              const float* __restrict__ m, float* __restrict__ den) {
  int e = blockIdx.x * blockDim.x + threadIdx.x;
  if (e >= E) return;
  int row = ei[e], col = ei[E + e];
  const float4 s4 = *(const float4*)(aS + (size_t)row * 4);
  const float4 d4 = *(const float4*)(aD + (size_t)col * 4);
  const float4 m4 = *(const float4*)(m  + (size_t)col * 4);
  float* dp = den + (size_t)col * 4;
  atomicAdd(dp + 0, __expf(lrelu(s4.x + d4.x, 0.2f) - m4.x));
  atomicAdd(dp + 1, __expf(lrelu(s4.y + d4.y, 0.2f) - m4.y));
  atomicAdd(dp + 2, __expf(lrelu(s4.z + d4.z, 0.2f) - m4.z));
  atomicAdd(dp + 3, __expf(lrelu(s4.w + d4.w, 0.2f) - m4.w));
}

// one wave (32 lanes) per edge; each lane scatters 2 of the 64 features
__global__ __launch_bounds__(256)
void hgat_edge_agg(const int* __restrict__ ei, int E,
                   const float* __restrict__ aS, const float* __restrict__ aD,
                   const float* __restrict__ m, const float* __restrict__ den,
                   const float* __restrict__ hs, float* __restrict__ out) {
  int e = blockIdx.x * 8 + (threadIdx.x >> 5);
  if (e >= E) return;
  const int lane = threadIdx.x & 31;
  const int row = ei[e], col = ei[E + e];
  #pragma unroll
  for (int it = 0; it < 2; ++it) {
    const int f = lane + 32 * it;
    const int h = f >> 4;
    const float eh = lrelu(aS[(size_t)row * 4 + h] + aD[(size_t)col * 4 + h], 0.2f);
    const float w  = __expf(eh - m[(size_t)col * 4 + h]) /
                     (den[(size_t)col * 4 + h] + 1e-16f);
    atomicAdd(out + (size_t)col * 64 + f, hs[(size_t)row * 64 + f] * w);
  }
}

__global__ void hgat_finalize(const float* __restrict__ out, float* __restrict__ xn,
                              float* __restrict__ sum, int n) {
  int i = blockIdx.x * blockDim.x + threadIdx.x;
  if (i < n) { float v = lrelu(out[i], 0.01f); xn[i] = v; if (sum) sum[i] += v; }
}

__global__ void hgat_scale_out(const float* __restrict__ s, float* __restrict__ o,
                               int n, float sc) {
  int i = blockIdx.x * blockDim.x + threadIdx.x;
  if (i < n) o[i] = s[i] * sc;
}

// ---------------------------------------------------------------------------
static inline int cdiv(int a, int b) { return (a + b - 1) / b; }

extern "C" void kernel_launch(void* const* d_in, const int* in_sizes, int n_in,
                              void* d_out, int out_size, void* d_ws, size_t ws_size,
                              hipStream_t stream) {
  (void)n_in; (void)out_size; (void)ws_size;
  const float* x_in[3] = { (const float*)d_in[0], (const float*)d_in[1], (const float*)d_in[2] };
  const float* Wsrc   = (const float*)d_in[3];
  const float* Wdst   = (const float*)d_in[4];
  const float* attS   = (const float*)d_in[5];
  const float* attD   = (const float*)d_in[6];
  const float* bias   = (const float*)d_in[7];
  const int* ei[8]; int Ecnt[8];
  for (int t = 0; t < 8; ++t) { ei[t] = (const int*)d_in[8 + t]; Ecnt[t] = in_sizes[8 + t] / 2; }

  static const int NT[3]   = { NU, NA, NC };
  static const int srcT[8] = { 0, 1, 0, 0, 1, 2, 0, 2 };
  static const int dstT[8] = { 1, 0, 0, 0, 2, 1, 2, 0 };

  // ---- workspace carve ---------------------------------------------------
  float* ws = (float*)d_ws;
  size_t off = 0;
  auto alloc = [&](size_t n) { float* p = ws + off; off += n; return p; };
  float* xbuf[3][2];
  for (int ty = 0; ty < 3; ++ty) {
    xbuf[ty][0] = alloc((size_t)NT[ty] * 64);
    xbuf[ty][1] = alloc((size_t)NT[ty] * 64);
  }
  float* sumU = alloc((size_t)NU * 64);
  float* sumA = alloc((size_t)NA * 64);
  float* hs   = alloc((size_t)NU * 64);    // max source count
  float* aS   = alloc((size_t)NU * 4);
  float* aD   = alloc((size_t)NU * 4);
  float* mbuf = alloc((size_t)NU * 4);
  float* den  = alloc((size_t)NU * 4);
  float* outb[3] = { alloc((size_t)NU * 64), alloc((size_t)NA * 64), alloc((size_t)NC * 64) };

  // ---- init: x0 copies + running sums ------------------------------------
  hgat_copy_init<<<cdiv(NU * 64, 256), 256, 0, stream>>>(x_in[0], xbuf[0][0], sumU, NU * 64);
  hgat_copy_init<<<cdiv(NA * 64, 256), 256, 0, stream>>>(x_in[1], xbuf[1][0], sumA, NA * 64);
  hgat_copy_init<<<cdiv(NC * 64, 256), 256, 0, stream>>>(x_in[2], xbuf[2][0], nullptr, NC * 64);

  int cur = 0;
  for (int l = 0; l < 2; ++l) {
    const int nxt = cur ^ 1;
    const float* biasL = bias + (size_t)l * 8 * 64;
    // out init = sum of biases of incoming edge types (HeteroConv 'sum')
    hgat_out_bias<<<cdiv(NU * 64, 256), 256, 0, stream>>>(outb[0], biasL, 1, 2, 3, 7, 4, NU * 64);
    hgat_out_bias<<<cdiv(NA * 64, 256), 256, 0, stream>>>(outb[1], biasL, 0, 5, 0, 0, 2, NA * 64);
    hgat_out_bias<<<cdiv(NC * 64, 256), 256, 0, stream>>>(outb[2], biasL, 4, 6, 0, 0, 2, NC * 64);

    for (int t = 0; t < 8; ++t) {
      const int s = srcT[t], d = dstT[t], E = Ecnt[t];
      const size_t wo = ((size_t)l * 8 + t) * 64 * 64;
      const size_t ao = ((size_t)l * 8 + t) * 64;   // H*C = 64
      hgat_xform<<<cdiv(NT[s], 64), 128, 0, stream>>>(xbuf[s][cur], Wsrc + wo, attS + ao,
                                                      hs, aS, NT[s], 1);
      hgat_xform<<<cdiv(NT[d], 64), 128, 0, stream>>>(xbuf[d][cur], Wdst + wo, attD + ao,
                                                      hs, aD, NT[d], 0);
      hgat_init_md<<<cdiv(NT[d] * 4, 256), 256, 0, stream>>>(mbuf, den, NT[d] * 4);
      hgat_edge_max<<<cdiv(E, 256), 256, 0, stream>>>(ei[t], E, aS, aD, mbuf);
      hgat_edge_den<<<cdiv(E, 256), 256, 0, stream>>>(ei[t], E, aS, aD, mbuf, den);
      hgat_edge_agg<<<cdiv(E, 8), 256, 0, stream>>>(ei[t], E, aS, aD, mbuf, den, hs, outb[d]);
    }

    hgat_finalize<<<cdiv(NU * 64, 256), 256, 0, stream>>>(outb[0], xbuf[0][nxt], sumU, NU * 64);
    hgat_finalize<<<cdiv(NA * 64, 256), 256, 0, stream>>>(outb[1], xbuf[1][nxt], sumA, NA * 64);
    hgat_finalize<<<cdiv(NC * 64, 256), 256, 0, stream>>>(outb[2], xbuf[2][nxt], nullptr, NC * 64);
    cur = nxt;
  }

  float* outp = (float*)d_out;
  hgat_scale_out<<<cdiv(NU * 64, 256), 256, 0, stream>>>(sumU, outp, NU * 64, 1.f / 3.f);
  hgat_scale_out<<<cdiv(NA * 64, 256), 256, 0, stream>>>(sumA, outp + (size_t)NU * 64, NA * 64, 1.f / 3.f);
}
